// Real_Gated_Linear_Recurrent_Unit_49787260895286
// MI455X (gfx1250) — compile-verified, matrix operand order checked by reading, not run on hardware
//
#include <hip/hip_runtime.h>
#include <cstdint>

#define BATCH 8
#define SEQ   2048
#define DDIM  512
#define HID   1536
#define MTOT  (BATCH * SEQ)   // 16384

typedef __bf16 v16bf __attribute__((ext_vector_type(16)));
typedef float  v8f   __attribute__((ext_vector_type(8)));

struct alignas(16) U128 { unsigned int a, b, c, d; };

union Frag { v16bf v; U128 q[2]; };

#define LDSS 40  // LDS row stride in uint16 units (80 B): conflict-free 16-row b128 reads

// Fragment loader: works for both A (row m, K-chunks) and B (col n of W^T = row n of W).
// Lane l: row = l&15, chunks k = {hh*8 .. hh*8+7} and {16+hh*8 ..} with hh = l>>4.
__device__ __forceinline__ v16bf load_frag_lds(const uint16_t* base, int lane) {
  const int r  = lane & 15;
  const int hh = lane >> 4;
  const uint16_t* p = base + r * LDSS + hh * 8;
  Frag f;
  f.q[0] = *(const U128*)(p);
  f.q[1] = *(const U128*)(p + 16);
  return f.v;
}

__device__ __forceinline__ v8f wmma_bf16(v16bf a, v16bf b, v8f c) {
  return __builtin_amdgcn_wmma_f32_16x16x32_bf16(false, a, false, b, (short)0, c,
                                                 false, false);
}

// CDNA5 async copy: global -> LDS, 16 bytes, tracked by ASYNCcnt (no VGPR data path).
// GV addressing: 64-bit per-lane global address, LDS byte offset in a VGPR.
__device__ __forceinline__ void async_copy_b128(unsigned int lds_off, const void* gaddr) {
  asm volatile("global_load_async_to_lds_b128 %0, %1, off"
               :
               : "v"(lds_off), "v"((unsigned long long)(uintptr_t)gaddr)
               : "memory");
}

__device__ __forceinline__ void wait_asynccnt0() {
  asm volatile("s_wait_asynccnt 0x0" ::: "memory");
}

// Split f32 -> bf16 hi + bf16 lo (RNE both); x ~= hi + lo to ~16 mantissa bits.
__global__ __launch_bounds__(256) void rglru_split_kernel(
    const float* __restrict__ src, uint16_t* __restrict__ hi,
    uint16_t* __restrict__ lo, int n) {
  int i = blockIdx.x * blockDim.x + threadIdx.x;
  int stride = gridDim.x * blockDim.x;
  for (; i < n; i += stride) {
    float x = src[i];
    unsigned int u  = __float_as_uint(x);
    unsigned int hr = u + 0x7FFFu + ((u >> 16) & 1u);
    unsigned short h16 = (unsigned short)(hr >> 16);
    float hf = __uint_as_float(((unsigned int)h16) << 16);
    float l  = x - hf;
    unsigned int ul = __float_as_uint(l);
    unsigned int lr = ul + 0x7FFFu + ((ul >> 16) & 1u);
    hi[i] = h16;
    lo[i] = (unsigned short)(lr >> 16);
  }
}

// Fused triple-GEMM (split-bf16) + gate epilogue.
// Block tile: 128(M) x 64(N); 8 waves as 4(M) x 2(N); wave tile 32x32.
// Double-buffered LDS fed by global_load_async_to_lds_b128.
__global__ __launch_bounds__(256) void rglru_gemm_kernel(
    const uint16_t* __restrict__ xh, const uint16_t* __restrict__ xl,
    const uint16_t* __restrict__ Wah, const uint16_t* __restrict__ Wal,
    const uint16_t* __restrict__ Wxh, const uint16_t* __restrict__ Wxl,
    const uint16_t* __restrict__ Wph, const uint16_t* __restrict__ Wpl,
    const float* __restrict__ ba, const float* __restrict__ bx,
    const float* __restrict__ Lam,
    float* __restrict__ Aout, float* __restrict__ Vout) {
  __shared__ uint16_t sXh[2][128 * LDSS];
  __shared__ uint16_t sXl[2][128 * LDSS];
  __shared__ uint16_t sWh[2][3 * 64 * LDSS];
  __shared__ uint16_t sWl[2][3 * 64 * LDSS];

  const int tid  = threadIdx.x;
  const int lane = tid & 31;
  const int wave = tid >> 5;
  const int wm   = wave & 3;   // wave row (M)
  const int wn   = wave >> 2;  // wave col (N)
  const int bm   = blockIdx.x * 128;
  const int bn   = blockIdx.y * 64;

  // Async stage of one 32-wide K slice into buffer `buf`.
  // 640 row-units of 64 B (4 x async b128 each) spread over 256 threads.
  auto stage = [&](int buf, int kk) {
    for (int u = tid; u < 640; u += 256) {
      const uint16_t* gsrc;
      uint16_t* ldst;
      if (u < 256) {
        int r = u & 127;
        bool isLo = (u >= 128);
        gsrc = (isLo ? xl : xh) + (size_t)(bm + r) * DDIM + kk;
        ldst = (isLo ? sXl[buf] : sXh[buf]) + r * LDSS;
      } else {
        int w   = u - 256;      // 0..383
        int mat = w >> 7;       // 0..2
        int rem = w & 127;
        bool isLo = (rem >= 64);
        int r = rem & 63;
        const uint16_t* wp;
        if (mat == 0)      wp = isLo ? Wal : Wah;
        else if (mat == 1) wp = isLo ? Wxl : Wxh;
        else               wp = isLo ? Wpl : Wph;
        gsrc = wp + (size_t)(bn + r) * DDIM + kk;
        ldst = (isLo ? sWl[buf] : sWh[buf]) + (mat * 64 + r) * LDSS;
      }
      const unsigned int loff = (unsigned int)(uintptr_t)ldst;  // LDS byte offset
      const char* g = (const char*)gsrc;                        // 64B-aligned
      async_copy_b128(loff +  0, g +  0);
      async_copy_b128(loff + 16, g + 16);
      async_copy_b128(loff + 32, g + 32);
      async_copy_b128(loff + 48, g + 48);
    }
  };

  v8f zero;
#pragma unroll
  for (int j = 0; j < 8; ++j) zero[j] = 0.0f;
  v8f acc[3][2][2];
#pragma unroll
  for (int m = 0; m < 3; ++m)
#pragma unroll
    for (int i = 0; i < 2; ++i)
#pragma unroll
      for (int j = 0; j < 2; ++j) acc[m][i][j] = zero;

  // Prologue: fill buffer 0.
  stage(0, 0);
  wait_asynccnt0();
  __syncthreads();

  for (int kk = 0; kk < DDIM; kk += 32) {
    const int cur = (kk >> 5) & 1;
    // Prefetch next K slice into the other buffer while computing on this one.
    if (kk + 32 < DDIM) stage(cur ^ 1, kk + 32);

    // ---- fragments + WMMA: acc += Xhi*Whi + Xlo*Whi + Xhi*Wlo ----
    const int xrow = wm * 32;
    v16bf axh0 = load_frag_lds(&sXh[cur][(xrow +  0) * LDSS], lane);
    v16bf axh1 = load_frag_lds(&sXh[cur][(xrow + 16) * LDSS], lane);
    v16bf axl0 = load_frag_lds(&sXl[cur][(xrow +  0) * LDSS], lane);
    v16bf axl1 = load_frag_lds(&sXl[cur][(xrow + 16) * LDSS], lane);

    v16bf bhf[6], blf[6];  // all 12 B fragments live -> parallel ds_load issue
#pragma unroll
    for (int mat = 0; mat < 3; ++mat) {
#pragma unroll
      for (int ni = 0; ni < 2; ++ni) {
        const int wrow = mat * 64 + wn * 32 + ni * 16;
        bhf[mat * 2 + ni] = load_frag_lds(&sWh[cur][wrow * LDSS], lane);
        blf[mat * 2 + ni] = load_frag_lds(&sWl[cur][wrow * LDSS], lane);
      }
    }
#pragma unroll
    for (int mat = 0; mat < 3; ++mat) {
#pragma unroll
      for (int ni = 0; ni < 2; ++ni) {
        const int q = mat * 2 + ni;
        acc[mat][0][ni] = wmma_bf16(axh0, bhf[q], acc[mat][0][ni]);
        acc[mat][0][ni] = wmma_bf16(axl0, bhf[q], acc[mat][0][ni]);
        acc[mat][0][ni] = wmma_bf16(axh0, blf[q], acc[mat][0][ni]);
        acc[mat][1][ni] = wmma_bf16(axh1, bhf[q], acc[mat][1][ni]);
        acc[mat][1][ni] = wmma_bf16(axl1, bhf[q], acc[mat][1][ni]);
        acc[mat][1][ni] = wmma_bf16(axh1, blf[q], acc[mat][1][ni]);
      }
    }

    wait_asynccnt0();   // this wave's prefetch into buf cur^1 has landed
    __syncthreads();    // all waves done reading cur + all prefetches visible
  }

  // ---- fused gate epilogue ----
  // C layout: VGPR j holds rows {j, j+8}; lanes 0-15 / 16-31 select the half;
  // column n = lane & 15.
  const int l15 = lane & 15;
  const int lhi = lane >> 4;
#pragma unroll
  for (int ni = 0; ni < 2; ++ni) {
    const int n = bn + wn * 32 + ni * 16 + l15;
    const float bav = ba[n];
    const float bxv = bx[n];
    const float lam = Lam[n];
    const float cn  = -8.0f * log1pf(__expf(-lam));  // -C * softplus(-Lambda)
#pragma unroll
    for (int mi = 0; mi < 2; ++mi) {
      const int mbase = bm + wm * 32 + mi * 16 + lhi * 8;
#pragma unroll
      for (int j = 0; j < 8; ++j) {
        const int m = mbase + j;
        const float ga = acc[0][mi][ni][j] + bav;
        const float gx = acc[1][mi][ni][j] + bxv;
        const float gp = acc[2][mi][ni][j];
        const float r  = 1.0f / (1.0f + __expf(-ga));
        const float ig = 1.0f / (1.0f + __expf(-gx));
        const float a  = __expf(cn * r);
        float w2 = 1.0f - a * a;
        w2 = fmaxf(w2, 1e-8f);
        const float v = sqrtf(w2) * (ig * gp);
        const size_t idx = (size_t)m * HID + n;
        Aout[idx] = a;
        Vout[idx] = v;
      }
    }
  }
}

// In-place linear recurrence over S: h_t = a_t*h_{t-1} + v_t (v lives in HV).
__global__ __launch_bounds__(256) void rglru_scan_kernel(
    const float* __restrict__ A, float* __restrict__ HV) {
  int t = blockIdx.x * blockDim.x + threadIdx.x;
  if (t >= BATCH * HID) return;
  const int b = t / HID;
  const int h = t % HID;
  const float* pa = A + (size_t)b * SEQ * HID + h;
  float* pv = HV + (size_t)b * SEQ * HID + h;
  float hc = 0.0f;
  for (int s = 0; s < SEQ; s += 8) {
    float av[8], vv[8];
#pragma unroll
    for (int j = 0; j < 8; ++j) {
      av[j] = pa[(size_t)j * HID];
      vv[j] = pv[(size_t)j * HID];
    }
#pragma unroll
    for (int j = 0; j < 8; ++j) {
      hc = fmaf(av[j], hc, vv[j]);
      pv[(size_t)j * HID] = hc;
    }
    pa += (size_t)8 * HID;
    pv += (size_t)8 * HID;
  }
}

extern "C" void kernel_launch(void* const* d_in, const int* in_sizes, int n_in,
                              void* d_out, int out_size, void* d_ws, size_t ws_size,
                              hipStream_t stream) {
  const float* x   = (const float*)d_in[0];
  const float* Wa  = (const float*)d_in[1];
  const float* Wx  = (const float*)d_in[2];
  const float* Wp  = (const float*)d_in[3];
  const float* ba  = (const float*)d_in[4];
  const float* bx  = (const float*)d_in[5];
  const float* Lam = (const float*)d_in[6];
  float* out = (float*)d_out;

  // Workspace layout (~144 MB total)
  char* w = (char*)d_ws;
  float* Aws = (float*)w;      w += (size_t)BATCH * SEQ * HID * sizeof(float);
  uint16_t* xh  = (uint16_t*)w; w += (size_t)BATCH * SEQ * DDIM * 2;
  uint16_t* xl  = (uint16_t*)w; w += (size_t)BATCH * SEQ * DDIM * 2;
  uint16_t* Wah = (uint16_t*)w; w += (size_t)HID * DDIM * 2;
  uint16_t* Wal = (uint16_t*)w; w += (size_t)HID * DDIM * 2;
  uint16_t* Wxh = (uint16_t*)w; w += (size_t)HID * DDIM * 2;
  uint16_t* Wxl = (uint16_t*)w; w += (size_t)HID * DDIM * 2;
  uint16_t* Wph = (uint16_t*)w; w += (size_t)HID * DDIM * 2;
  uint16_t* Wpl = (uint16_t*)w; w += (size_t)HID * DDIM * 2;

  rglru_split_kernel<<<2048, 256, 0, stream>>>(x, xh, xl, BATCH * SEQ * DDIM);
  rglru_split_kernel<<<768, 256, 0, stream>>>(Wa, Wah, Wal, HID * DDIM);
  rglru_split_kernel<<<768, 256, 0, stream>>>(Wx, Wxh, Wxl, HID * DDIM);
  rglru_split_kernel<<<768, 256, 0, stream>>>(Wp, Wph, Wpl, HID * DDIM);

  dim3 grid(MTOT / 128, HID / 64);  // (128, 24)
  rglru_gemm_kernel<<<grid, 256, 0, stream>>>(xh, xl, Wah, Wal, Wxh, Wxl, Wph, Wpl,
                                              ba, bx, Lam, Aws, out);

  rglru_scan_kernel<<<(BATCH * HID + 255) / 256, 256, 0, stream>>>(Aws, out);
}
